// a3c_lstm_ga_concat_gavector_50929722196564
// MI455X (gfx1250) — compile-verified, matrix operand order checked by reading, not run on hardware
//
#include <hip/hip_runtime.h>
#include <hip/hip_bf16.h>
#include <math.h>

// ---------- types ----------
typedef _Float16 f16;
typedef __attribute__((ext_vector_type(16))) _Float16 v16h;
typedef __attribute__((ext_vector_type(8)))  float    v8f;
typedef __attribute__((ext_vector_type(4)))  unsigned u32x4;
typedef __attribute__((ext_vector_type(8)))  int      i32x8;
typedef __attribute__((ext_vector_type(4)))  int      i32x4;

union AFrag { v16h v; unsigned u[8]; };
union CFrag { v8f  v; float    f[8]; };

#if defined(__has_builtin)
#if __has_builtin(__builtin_amdgcn_tensor_load_to_lds)
#define HAVE_TDM 1
#endif
#endif
#ifndef HAVE_TDM
#define HAVE_TDM 0
#endif

#if __has_include(<hip/amd_detail/amd_gfx1250_TDM.h>)
#define TDM_SIX_ARG 1
#else
#define TDM_SIX_ARG 0
#endif

__device__ __forceinline__ float sigm(float x) { return 1.0f / (1.0f + __expf(-x)); }

__device__ __forceinline__ unsigned pack2h(float a, float b) {
  union { f16 h[2]; unsigned u; } p;
  p.h[0] = (f16)a; p.h[1] = (f16)b;
  return p.u;
}

// K-local offset inside a 16x32 f16 A/B WMMA fragment for VGPR v, lane-half hi(0/8)
// (ISA 7.12.2: v0..3 hold K pairs {0..7}/{8..15}, v4..7 hold {16..23}/{24..31})
__device__ __forceinline__ int klocal(int v, int hi) {
  return ((v < 4) ? (2 * v) : (16 + 2 * (v - 4))) + hi;
}

// raw 32-bit LDS byte offset of a __shared__ object (generic -> addrspace(3) -> int)
__device__ __forceinline__ unsigned lds_offset_of(const void* p) {
  return (unsigned)(unsigned long long)(__attribute__((address_space(3))) const void*)p;
}

// Issue one TDM 2D tile load: 3 rows x 768 f16, row stride 2048*768 elements,
// i.e. the gi[s][step][0..767] slice for s=0..2, into LDS at lds_off.
__device__ __forceinline__ void tdm_load_gi_tile(const f16* gsrc, unsigned lds_off) {
#if HAVE_TDM
  unsigned long long ga = (unsigned long long)gsrc;
  u32x4 g0;
  g0.x = 1u;                                                // count=1 (valid), user mode
  g0.y = lds_off;                                           // lds_addr
  g0.z = (unsigned)(ga & 0xFFFFFFFFu);                      // global_addr[31:0]
  g0.w = (unsigned)((ga >> 32) & 0x01FFFFFFu) | (2u << 30); // global_addr[56:32] | type=2
  i32x8 g1;
  g1[0] = 0x00010000;            // workgroup_mask=0, data_size=1 (2 bytes)
  g1[1] = (int)(768u << 16);     // tensor_dim0[15:0]=768 at bits[63:48]
  g1[2] = (int)(3u << 16);       // tensor_dim0[31:16]=0 | tensor_dim1[15:0]=3
  g1[3] = (int)(768u << 16);     // tensor_dim1[31:16]=0 | tile_dim0=768
  g1[4] = 3;                     // tile_dim1=3, tile_dim2=0
  g1[5] = (int)(2048u * 768u);   // tensor_dim0_stride[31:0] (elements)
  g1[6] = 0;                     // stride hi | tensor_dim1_stride lo
  g1[7] = 0;
  i32x4 z4 = {0, 0, 0, 0};
#if TDM_SIX_ARG
  i32x8 z8 = {0, 0, 0, 0, 0, 0, 0, 0};
  __builtin_amdgcn_tensor_load_to_lds(g0, g1, z4, z4, z8, 0);
#else
  __builtin_amdgcn_tensor_load_to_lds(g0, g1, z4, z4, 0);
#endif
#else
  (void)gsrc; (void)lds_off;
#endif
}

// ---------- 1) weight packing: f32 [N][K] -> u32 pairs [K/2][N] (f16,f16) ----------
__global__ void pack_pairs_kernel(const float* __restrict__ w, unsigned* __restrict__ out,
                                  int kpairs, int N, int ldw) {
  int i = blockIdx.x * blockDim.x + threadIdx.x;
  if (i >= kpairs * N) return;
  int n = i % N, kk = i / N;
  out[i] = pack2h(w[(size_t)n * ldw + 2 * kk], w[(size_t)n * ldw + 2 * kk + 1]);
}

// ---------- 2) direct conv + relu (f32, VALID, square kernel) ----------
__global__ void conv_relu_kernel(const float* __restrict__ in, const float* __restrict__ w,
                                 const float* __restrict__ b, float* __restrict__ out,
                                 int Cin, int Hin, int Win, int Cout, int K, int S,
                                 int Hout, int Wout) {
  int i = blockIdx.x * blockDim.x + threadIdx.x;
  int total = Cout * Hout * Wout;
  if (i >= total) return;
  int ox = i % Wout, oy = (i / Wout) % Hout, oc = i / (Wout * Hout);
  float acc = b[oc];
  const float* wp = w + (size_t)oc * Cin * K * K;
  for (int ic = 0; ic < Cin; ic++) {
    const float* ip = in + ((size_t)ic * Hin + oy * S) * Win + ox * S;
    for (int ky = 0; ky < K; ky++)
      for (int kx = 0; kx < K; kx++)
        acc = fmaf(ip[ky * Win + kx], wp[((size_t)ic * K + ky) * K + kx], acc);
  }
  out[i] = fmaxf(acc, 0.0f);
}

// ---------- 3) gi precompute: [6144 x 32] @ [32 x 768] with WMMA f16 ----------
__global__ __launch_bounds__(256) void gi_gemm_kernel(
    const int* __restrict__ curr, const int* __restrict__ prev, const int* __restrict__ next,
    const float* __restrict__ emb, const unsigned* __restrict__ Bih,
    const float* __restrict__ b_ih, f16* __restrict__ gi) {
  __shared__ f16 As[16][34];
  const int m0 = blockIdx.x * 16;
  const int t = threadIdx.x;

  for (int idx = t; idx < 16 * 32; idx += 256) {
    int ml = idx >> 5, e = idx & 31;
    int m = m0 + ml;
    int s = m >> 11, l = m & 2047;
    const int* tokp = (s == 0) ? curr : ((s == 1) ? prev : next);
    As[ml][e] = (f16)emb[(size_t)tokp[l] * 32 + e];
  }
  __syncthreads();

  const int lane = t & 31, w = t >> 5;
  const int mloc = lane & 15;
  const int hi = (lane & 16) >> 1;

  AFrag a;
  const unsigned* arow = (const unsigned*)&As[mloc][0];
#pragma unroll
  for (int v = 0; v < 8; v++) a.u[v] = arow[klocal(v, hi) >> 1];

  for (int i = 0; i < 6; i++) {
    int tile = w * 6 + i;
    int n = tile * 16 + (lane & 15);
    AFrag bf;
#pragma unroll
    for (int v = 0; v < 8; v++) bf.u[v] = Bih[(size_t)(klocal(v, hi) >> 1) * 768 + n];

    CFrag c;
#pragma unroll
    for (int v = 0; v < 8; v++) c.f[v] = 0.0f;
    c.v = __builtin_amdgcn_wmma_f32_16x16x32_f16(false, a.v, false, bf.v,
                                                 (short)0, c.v, false, false);
    float bn = b_ih[n];
#pragma unroll
    for (int v = 0; v < 8; v++) {
      int M = v + hi;
      gi[(size_t)(m0 + M) * 768 + n] = (f16)(c.f[v] + bn);
    }
  }
}

// ---------- 4) GRU scan: 2048 serial steps, W_hh register-resident, WMMA per step,
//             gi slices streamed one step ahead via the Tensor Data Mover ----------
__global__ __launch_bounds__(768, 1) void gru_scan_kernel(
    const unsigned* __restrict__ Bhh, const f16* __restrict__ gi,
    const float* __restrict__ b_hh, float* __restrict__ h3out) {
  __shared__ f16 Ah[16][258];        // A matrix (rows 0..2 = h), padded, conflict-free
  __shared__ float GH[3 * 768];      // h @ W_hh^T results
#if HAVE_TDM
  __shared__ f16 GIbuf[2][3 * 768];  // TDM double buffer for gi[s][step][:]
#endif

  const int t = threadIdx.x, lane = t & 31, w = t >> 5;
  const int mloc = lane & 15;
  const int hi = (lane & 16) >> 1;

  {
    f16* ap = &Ah[0][0];
    for (int i = t; i < 16 * 258; i += 768) ap[i] = (f16)0.0f;
  }

  // register-resident W_hh fragments: 2 tiles x 8 K-chunks x 8 VGPRs
  AFrag b[2][8];
#pragma unroll
  for (int tt = 0; tt < 2; tt++) {
    int n = (2 * w + tt) * 16 + (lane & 15);
#pragma unroll
    for (int c = 0; c < 8; c++) {
#pragma unroll
      for (int v = 0; v < 8; v++) {
        int k = 32 * c + klocal(v, hi);
        b[tt][c].u[v] = Bhh[(size_t)(k >> 1) * 768 + n];
      }
    }
  }

  const int s = t / 256, j = t % 256;
  float h = 0.0f;
  const float bhr = b_hh[j], bhz = b_hh[256 + j], bhn = b_hh[512 + j];
#if !HAVE_TDM
  const f16* gip = gi + (size_t)s * 2048 * 768 + j;
#endif

#if HAVE_TDM
  if (w == 0) tdm_load_gi_tile(gi, lds_offset_of(&GIbuf[0][0]));  // prefetch step 0
#endif
  __syncthreads();

  for (int step = 0; step < 2048; step++) {
    // ---- matrix phase: GH = h_rows @ W_hh^T ----
    CFrag c0, c1;
#pragma unroll
    for (int v = 0; v < 8; v++) { c0.f[v] = 0.0f; c1.f[v] = 0.0f; }

    const unsigned* arow = (const unsigned*)&Ah[mloc][0];
#pragma unroll
    for (int c = 0; c < 8; c++) {
      AFrag a;
#pragma unroll
      for (int v = 0; v < 8; v++) a.u[v] = arow[(32 * c + klocal(v, hi)) >> 1];
      c0.v = __builtin_amdgcn_wmma_f32_16x16x32_f16(false, a.v, false, b[0][c].v,
                                                    (short)0, c0.v, false, false);
      c1.v = __builtin_amdgcn_wmma_f32_16x16x32_f16(false, a.v, false, b[1][c].v,
                                                    (short)0, c1.v, false, false);
    }
#pragma unroll
    for (int v = 0; v < 8; v++) {
      int M = v + hi;
      if (M < 3) {
        GH[M * 768 + (2 * w + 0) * 16 + (lane & 15)] = c0.f[v];
        GH[M * 768 + (2 * w + 1) * 16 + (lane & 15)] = c1.f[v];
      }
    }

#if HAVE_TDM
    // step's gi tile must be resident; then launch step+1 into the other buffer.
    // The in-flight DMA always targets the buffer NOT read this step.
    if (w == 0) {
      __builtin_amdgcn_s_wait_tensorcnt(0);
      if (step + 1 < 2048)
        tdm_load_gi_tile(gi + (size_t)(step + 1) * 768,
                         lds_offset_of(&GIbuf[(step + 1) & 1][0]));
    }
#endif
    __syncthreads();

    // ---- gate phase: 3 x 256 elementwise GRU update ----
#if HAVE_TDM
    const f16* g = &GIbuf[step & 1][0];
    float i_r = (float)g[s * 768 + j];
    float i_z = (float)g[s * 768 + 256 + j];
    float i_n = (float)g[s * 768 + 512 + j];
#else
    const f16* g = gip + (size_t)step * 768;
    __builtin_prefetch(g + 768, 0, 1);
    float i_r = (float)g[0], i_z = (float)g[256], i_n = (float)g[512];
#endif
    float h_r = GH[s * 768 + j] + bhr;
    float h_z = GH[s * 768 + 256 + j] + bhz;
    float h_n = GH[s * 768 + 512 + j] + bhn;
    float r = sigm(i_r + h_r);
    float z = sigm(i_z + h_z);
    float n = tanhf(i_n + r * h_n);
    h = (1.0f - z) * n + z * h;
    Ah[s][j] = (f16)h;
    __syncthreads();
  }
  h3out[s * 256 + j] = h;
}

// ---------- 5) attention (bug-faithful: only attn[0] row is ever used) ----------
__global__ void attn0_kernel(const float* __restrict__ h3, const float* __restrict__ attn_w,
                             const float* __restrict__ attn_b, float* __restrict__ attn0) {
  int jj = threadIdx.x;
  if (jj >= 64) return;
  float acc = attn_b[jj];
  for (int k = 0; k < 256; k++) acc = fmaf(h3[k], attn_w[(size_t)jj * 256 + k], acc);
  attn0[jj] = sigm(acc);
}

// ---------- 6) image fusion + linear 2304->256 + relu ----------
__global__ void feat_kernel(const float* __restrict__ conv3o, const float* __restrict__ attn0,
                            const float* __restrict__ lin_w, const float* __restrict__ lin_b,
                            float* __restrict__ feat) {
  int o = blockIdx.x * blockDim.x + threadIdx.x;
  if (o >= 256) return;
  float acc = lin_b[o];
  const float* wrow = lin_w + (size_t)o * 2304;
  for (int c = 0; c < 64; c++) {
    float ac = attn0[c];
    for (int p = 0; p < 36; p++) {
      int i = c * 36 + p;
      acc = fmaf(conv3o[i] * ac, wrow[i], acc);
    }
  }
  feat[o] = fmaxf(acc, 0.0f);
}

// ---------- 7) LSTM gates: 3072 x (768 + 768) matvec ----------
__global__ void lstm_gates_kernel(const float* __restrict__ feat, const float* __restrict__ hx,
                                  const float* __restrict__ w_ih, const float* __restrict__ w_hh,
                                  const float* __restrict__ b_ih, const float* __restrict__ b_hh,
                                  float* __restrict__ gates) {
  int gidx = blockIdx.x * blockDim.x + threadIdx.x;
  if (gidx >= 3072) return;
  float acc = b_ih[gidx] + b_hh[gidx];
  const float* wi = w_ih + (size_t)gidx * 768;
  const float* wh = w_hh + (size_t)gidx * 768;
  for (int k = 0; k < 768; k++) acc = fmaf(feat[k & 255], wi[k], acc);
  for (int k = 0; k < 768; k++) acc = fmaf(hx[k], wh[k], acc);
  gates[gidx] = acc;
}

// ---------- 8) LSTM cell elementwise; writes hx_new/cx_new into d_out ----------
__global__ void lstm_cell_kernel(const float* __restrict__ gates, const float* __restrict__ cx,
                                 float* __restrict__ out) {
  int jj = blockIdx.x * blockDim.x + threadIdx.x;
  if (jj >= 768) return;
  float gi_ = gates[jj], gf = gates[768 + jj], gg = gates[1536 + jj], go = gates[2304 + jj];
  float c = sigm(gf) * cx[jj] + sigm(gi_) * tanhf(gg);
  float hh = sigm(go) * tanhf(c);
  out[5 + jj] = hh;
  out[773 + jj] = c;
}

// ---------- 9) heads over z = [hx_new(768), time_emb[tx](32)] ----------
__global__ void heads_kernel(const float* __restrict__ outbuf, const float* __restrict__ time_emb,
                             const int* __restrict__ tx,
                             const float* __restrict__ critic_w, const float* __restrict__ critic_b,
                             const float* __restrict__ actor_w, const float* __restrict__ actor_b,
                             float* __restrict__ out) {
  int t = threadIdx.x;
  if (t >= 5) return;
  const float* wrow = (t == 0) ? critic_w : (actor_w + (size_t)(t - 1) * 800);
  float acc = (t == 0) ? critic_b[0] : actor_b[t - 1];
  const float* te = time_emb + (size_t)tx[0] * 32;
  for (int k = 0; k < 768; k++) acc = fmaf(outbuf[5 + k], wrow[k], acc);
  for (int k = 0; k < 32; k++) acc = fmaf(te[k], wrow[768 + k], acc);
  out[t] = acc;
}

// ---------- launcher ----------
extern "C" void kernel_launch(void* const* d_in, const int* in_sizes, int n_in,
                              void* d_out, int out_size, void* d_ws, size_t ws_size,
                              hipStream_t stream) {
  (void)in_sizes; (void)n_in; (void)out_size; (void)ws_size;

  const float* x        = (const float*)d_in[0];
  const int*   curr     = (const int*)d_in[1];
  const int*   prev     = (const int*)d_in[2];
  const int*   nxt      = (const int*)d_in[3];
  const int*   tx       = (const int*)d_in[4];
  const float* hx       = (const float*)d_in[5];
  const float* cx       = (const float*)d_in[6];
  const float* conv1_w  = (const float*)d_in[7];
  const float* conv1_b  = (const float*)d_in[8];
  const float* conv2_w  = (const float*)d_in[9];
  const float* conv2_b  = (const float*)d_in[10];
  const float* conv3_w  = (const float*)d_in[11];
  const float* conv3_b  = (const float*)d_in[12];
  const float* emb      = (const float*)d_in[13];
  const float* time_emb = (const float*)d_in[14];
  const float* gru_w_ih = (const float*)d_in[15];
  const float* gru_w_hh = (const float*)d_in[16];
  const float* gru_b_ih = (const float*)d_in[17];
  const float* gru_b_hh = (const float*)d_in[18];
  const float* attn_w   = (const float*)d_in[19];
  const float* attn_b   = (const float*)d_in[20];
  const float* lin_w    = (const float*)d_in[21];
  const float* lin_b    = (const float*)d_in[22];
  const float* lstm_w_ih = (const float*)d_in[23];
  const float* lstm_w_hh = (const float*)d_in[24];
  const float* lstm_b_ih = (const float*)d_in[25];
  const float* lstm_b_hh = (const float*)d_in[26];
  const float* critic_w = (const float*)d_in[27];
  const float* critic_b = (const float*)d_in[28];
  const float* actor_w  = (const float*)d_in[29];
  const float* actor_b  = (const float*)d_in[30];

  float* out = (float*)d_out;
  char* ws = (char*)d_ws;

  // workspace layout (bytes, all 256-aligned)
  float*    conv1o = (float*)(ws + 0);                  // 128*30*30
  float*    conv2o = (float*)(ws + 460800);             // 64*14*14
  float*    conv3o = (float*)(ws + 510976);             // 64*6*6
  unsigned* Bih    = (unsigned*)(ws + 520192);          // 16*768
  unsigned* Bhh    = (unsigned*)(ws + 569344);          // 128*768
  f16*      gi     = (f16*)(ws + 962560);               // 6144*768 f16
  float*    h3     = (float*)(ws + 10399744);           // 3*256
  float*    attn0  = (float*)(ws + 10402816);           // 64
  float*    feat   = (float*)(ws + 10403072);           // 256
  float*    gates  = (float*)(ws + 10404096);           // 3072

  pack_pairs_kernel<<<(16 * 768 + 255) / 256, 256, 0, stream>>>(gru_w_ih, Bih, 16, 768, 32);
  pack_pairs_kernel<<<(128 * 768 + 255) / 256, 256, 0, stream>>>(gru_w_hh, Bhh, 128, 768, 256);

  conv_relu_kernel<<<(115200 + 255) / 256, 256, 0, stream>>>(x, conv1_w, conv1_b, conv1o,
                                                             3, 124, 124, 128, 8, 4, 30, 30);
  conv_relu_kernel<<<(12544 + 255) / 256, 256, 0, stream>>>(conv1o, conv2_w, conv2_b, conv2o,
                                                            128, 30, 30, 64, 4, 2, 14, 14);
  conv_relu_kernel<<<(2304 + 255) / 256, 256, 0, stream>>>(conv2o, conv3_w, conv3_b, conv3o,
                                                           64, 14, 14, 64, 4, 2, 6, 6);

  gi_gemm_kernel<<<384, 256, 0, stream>>>(curr, prev, nxt, emb, Bih, gru_b_ih, gi);
  gru_scan_kernel<<<1, 768, 0, stream>>>(Bhh, gi, gru_b_hh, h3);

  attn0_kernel<<<1, 64, 0, stream>>>(h3, attn_w, attn_b, attn0);
  feat_kernel<<<1, 256, 0, stream>>>(conv3o, attn0, lin_w, lin_b, feat);
  lstm_gates_kernel<<<12, 256, 0, stream>>>(feat, hx, lstm_w_ih, lstm_w_hh,
                                            lstm_b_ih, lstm_b_hh, gates);
  lstm_cell_kernel<<<3, 256, 0, stream>>>(gates, cx, out);
  heads_kernel<<<1, 32, 0, stream>>>(out, time_emb, tx, critic_w, critic_b,
                                     actor_w, actor_b, out);
}
